// PrunedMultiHeadAttention_48344151883971
// MI455X (gfx1250) — matrix operand order repaired and checked by reading, not compiled, not run on hardware
//
#include <hip/hip_runtime.h>
#include <hip/hip_bf16.h>

typedef __attribute__((ext_vector_type(16))) _Float16 v16h;
typedef __attribute__((ext_vector_type(8)))  _Float16 v8h;
typedef __attribute__((ext_vector_type(8)))  float    v8f;

#define HIDDEN   1024
#define HEADS    16
#define HEAD_DIM 64
#define SEQ      2048
#define BATCH    2
#define NROWS    (BATCH * SEQ)   // 4096 token rows

// ---------------------------------------------------------------------------
// Fragment loaders (WMMA 16-bit operands in this file's contiguous K order, wave32)
//
// A 16x32 (MxK): lane L holds row M = L%16; elem j -> K = 2*khalf+j,
//                khalf = 8*(L/16) (16 contiguous K values).
// B 32x16 (KxN): lane L holds col N = L%16; elem j -> K = 16*(L/16)+j
//                (16 contiguous K values).
// C/D 16x16 f32: lane L holds col N = L%16; vgpr v -> row M = v + 8*(L/16).
// ---------------------------------------------------------------------------
__device__ __forceinline__ v16h load_afrag(const _Float16* base, int khalf) {
  v8h lo = *(const v8h*)(base + 2 * khalf);
  v8h hi = *(const v8h*)(base + 2 * khalf + 8);
  v16h r;
#pragma unroll
  for (int i = 0; i < 8; ++i) { r[i] = lo[i]; r[i + 8] = hi[i]; }
  return r;
}

__device__ __forceinline__ v16h load_bfrag(const _Float16* p) {
  v8h lo = *(const v8h*)(p);
  v8h hi = *(const v8h*)(p + 8);
  v16h r;
#pragma unroll
  for (int i = 0; i < 8; ++i) { r[i] = lo[i]; r[i + 8] = hi[i]; }
  return r;
}

// CDNA5 async copy global -> LDS (ASYNCcnt-tracked, 16B per lane).
// Operand order per ISA: VDST = per-lane LDS byte offset, VADDR = 64-bit addr.
__device__ __forceinline__ void async_copy_b128(unsigned lds_off, const _Float16* g) {
  unsigned long long ga = (unsigned long long)(size_t)g;
  asm volatile("global_load_async_to_lds_b128 %0, %1, off"
               :: "v"(lds_off), "v"(ga) : "memory");
}

// ---------------------------------------------------------------------------
// Elementwise prep kernels
// ---------------------------------------------------------------------------
__global__ void cvt_f32_f16(const float* __restrict__ in, _Float16* __restrict__ out, int n) {
  int i = blockIdx.x * blockDim.x + threadIdx.x;
  if (i < n) out[i] = (_Float16)in[i];
}

// pw[o,d] = (sigmoid(scores[o/32, d/32]) > 0.05) ? w[o,d] : 0, cast to f16
__global__ void prep_weight(const float* __restrict__ w, const float* __restrict__ sc,
                            _Float16* __restrict__ out) {
  int i = blockIdx.x * blockDim.x + threadIdx.x;   // 0 .. 1048575
  int o = i >> 10, d = i & 1023;
  float s   = sc[(o >> 5) * 32 + (d >> 5)];
  float sig = 1.0f / (1.0f + __expf(-s));
  out[i] = (sig > 0.05f) ? (_Float16)w[i] : (_Float16)0.0f;
}

// Bit-pack the attention mask: pk[q*64 + key/32] bit j = (mask[q, key32+j] != 0).
// One wave32 covers exactly one 32-key word via __ballot.
__global__ void pack_mask(const int* __restrict__ m, unsigned* __restrict__ pk) {
  int t = blockIdx.x * blockDim.x + threadIdx.x;   // 0 .. 4194303 = q*2048+key
  unsigned long long b = __ballot(m[t] != 0);
  if ((threadIdx.x & 31) == 0) pk[t >> 5] = (unsigned)b;
}

// v16 [b*S, H*D] -> vT [b, h, d, s]  (so PV B-frags are contiguous over keys)
__global__ void transpose_v(const _Float16* __restrict__ v, _Float16* __restrict__ vT) {
  int i = blockIdx.x * blockDim.x + threadIdx.x;   // 0 .. 4194303
  int row = i >> 10;            // b*S + s
  int c   = i & 1023;           // h*64 + d
  int b = row >> 11, s = row & 2047;
  vT[((size_t)(b * HIDDEN) + c) * SEQ + s] = v[i];
}

// ---------------------------------------------------------------------------
// WMMA GEMM: Y[M,N] = A[M,K] @ W[N,K]^T (+bias).  K = 1024.
// 256-thread blocks (8 waves). All 8 waves share one 64x32 B tile, async-
// copied into LDS (double buffered, ASYNCcnt). Each wave owns a 16x64 C tile.
// Per k-step: 1 async b128 issue/thread, 2 global b128 (A) + 8 ds b128 (B)
// per lane, 4 WMMAs -> global B traffic cut 8x vs direct loads.
// ---------------------------------------------------------------------------
template <bool F32OUT>
__global__ __launch_bounds__(256) void gemm_wmma(const _Float16* __restrict__ A,
                                                 const _Float16* __restrict__ W,
                                                 const float* __restrict__ bias,
                                                 void* __restrict__ outp, int M, int N) {
  const int K = HIDDEN;
  __shared__ __attribute__((aligned(16))) _Float16 shB[2][64 * 32];

  int tid  = threadIdx.x;
  int wv   = tid >> 5;
  int lane = tid & 31;
  int lo16 = lane & 15;
  int hi   = lane >> 4;
  int khalf = hi << 3;   // A-frag K split
  int kq    = hi << 4;   // B-frag K base

  int nblocks = N >> 6;
  int bm = (int)blockIdx.x / nblocks;
  int bn = (int)blockIdx.x % nblocks;
  int mbase = (bm << 7) + (wv << 4);   // 128 rows per block, 16 per wave
  int nbase = bn << 6;

  // Cooperative B staging: thread -> (row r in [0,64), 16B chunk c in [0,4))
  int r = tid >> 2, c = tid & 3;
  const _Float16* gB = W + (size_t)(nbase + r) * K + (c << 3);
  unsigned ldsB0 = (unsigned)(size_t)(&shB[0][r * 32 + (c << 3)]);
  unsigned ldsB1 = (unsigned)(size_t)(&shB[1][r * 32 + (c << 3)]);

  async_copy_b128(ldsB0, gB);        // kb = 0  -> buf 0
  async_copy_b128(ldsB1, gB + 32);   // kb = 32 -> buf 1

  v8f acc[4];
#pragma unroll
  for (int t = 0; t < 4; ++t)
#pragma unroll
    for (int i = 0; i < 8; ++i) acc[t][i] = 0.0f;

  const _Float16* aRow = A + (size_t)(mbase + lo16) * K;

  for (int kb = 0; kb < K; kb += 32) {
    int buf = (kb >> 5) & 1;
    // Async loads complete in order: <=1 outstanding means current buf landed.
    if (kb + 32 < K) asm volatile("s_wait_asynccnt 0x1" ::: "memory");
    else             asm volatile("s_wait_asynccnt 0x0" ::: "memory");
    __syncthreads();

    v16h af = load_afrag(aRow + kb, khalf);
    const _Float16* bb = &shB[buf][0];
#pragma unroll
    for (int t = 0; t < 4; ++t) {
      v16h bf = load_bfrag(bb + (t * 16 + lo16) * 32 + kq);
      acc[t] = __builtin_amdgcn_wmma_f32_16x16x32_f16(false, af, false, bf,
                                                      (short)0, acc[t], false, false);
    }
    __syncthreads();   // all waves done reading buf before it is overwritten
    if (kb + 64 < K) async_copy_b128(buf ? ldsB1 : ldsB0, gB + kb + 64);
  }

#pragma unroll
  for (int t = 0; t < 4; ++t) {
    int col  = nbase + t * 16 + lo16;
    float bv = 0.0f;
    if (F32OUT) bv = bias[col];
#pragma unroll
    for (int v = 0; v < 8; ++v) {
      int row = mbase + v + (hi << 3);
      if (F32OUT)
        ((float*)outp)[(size_t)row * N + col] = acc[t][v] + bv;
      else
        ((_Float16*)outp)[(size_t)row * N + col] = (_Float16)acc[t][v];
    }
  }
}

// ---------------------------------------------------------------------------
// Flash-style attention. One wave32 workgroup per 16-query tile per (b,h).
// S-tile = Q(16x64) @ K^T via 4 WMMAs per 32-key block; online softmax with
// __shfl_xor row reductions (rows live in 16-lane halves); P bounced
// C-layout -> A-layout through LDS; O += P(16x32) @ V(32x64) via 4 WMMAs
// against pre-transposed V^T. Mask comes from packed bitwords (1 u32 / row /
// 32-key block instead of 32 ints).
// ---------------------------------------------------------------------------
__global__ __launch_bounds__(32) void attn_wmma(const _Float16* __restrict__ q16,
                                                const _Float16* __restrict__ k16,
                                                const _Float16* __restrict__ vT,
                                                const unsigned* __restrict__ mpk,
                                                _Float16* __restrict__ aout) {
  __shared__ __attribute__((aligned(16))) _Float16 pshm[16 * 32];

  int id = blockIdx.x;
  int qt = id & 127;            // S/16 = 128 query tiles
  int h  = (id >> 7) & 15;
  int b  = id >> 11;
  int lane  = threadIdx.x;
  int lo16  = lane & 15, hi = lane >> 4;
  int khalf = hi << 3, kq = hi << 4;
  int qbase = qt << 4;
  const float scale = 0.125f;   // 1/sqrt(64)

  const _Float16* qrow = q16 + (size_t)(b * SEQ + qbase + lo16) * HIDDEN + h * HEAD_DIM;
  v16h qa0 = load_afrag(qrow, khalf);        // head dims 0..31
  v16h qa1 = load_afrag(qrow + 32, khalf);   // head dims 32..63

  v8f acc[4];
  float mrow[8], lrow[8];
#pragma unroll
  for (int v = 0; v < 8; ++v) { mrow[v] = -3.0e38f; lrow[v] = 0.0f; }
#pragma unroll
  for (int t = 0; t < 4; ++t)
#pragma unroll
    for (int i = 0; i < 8; ++i) acc[t][i] = 0.0f;

  for (int kb = 0; kb < SEQ; kb += 32) {
    // ---- scores: c0 = keys kb..kb+15, c1 = keys kb+16..kb+31
    const _Float16* kr0 = k16 + (size_t)(b * SEQ + kb + lo16) * HIDDEN + h * HEAD_DIM + kq;
    const _Float16* kr1 = kr0 + (size_t)16 * HIDDEN;
    v8f c0, c1;
#pragma unroll
    for (int i = 0; i < 8; ++i) { c0[i] = 0.0f; c1[i] = 0.0f; }
    c0 = __builtin_amdgcn_wmma_f32_16x16x32_f16(false, qa0, false, load_bfrag(kr0),      (short)0, c0, false, false);
    c0 = __builtin_amdgcn_wmma_f32_16x16x32_f16(false, qa1, false, load_bfrag(kr0 + 32), (short)0, c0, false, false);
    c1 = __builtin_amdgcn_wmma_f32_16x16x32_f16(false, qa0, false, load_bfrag(kr1),      (short)0, c1, false, false);
    c1 = __builtin_amdgcn_wmma_f32_16x16x32_f16(false, qa1, false, load_bfrag(kr1 + 32), (short)0, c1, false, false);

    // ---- online softmax (rows live within a 16-lane half: xor 1,2,4,8)
    float p0[8], p1[8];
#pragma unroll
    for (int v = 0; v < 8; ++v) {
      int qr = qbase + v + (hi << 3);
      unsigned wbits = mpk[(size_t)qr * (SEQ / 32) + (kb >> 5)];
      float s0 = ((wbits >> lo16) & 1u)        ? c0[v] * scale : -3.0e38f;
      float s1 = ((wbits >> (lo16 + 16)) & 1u) ? c1[v] * scale : -3.0e38f;
      float t = fmaxf(s0, s1);
#pragma unroll
      for (int d = 1; d < 16; d <<= 1) t = fmaxf(t, __shfl_xor(t, d, 32));
      float mn    = fmaxf(mrow[v], t);
      float alpha = __expf(mrow[v] - mn);
      mrow[v] = mn;
      float e0 = __expf(s0 - mn);
      float e1 = __expf(s1 - mn);
      float rsum = e0 + e1;
#pragma unroll
      for (int d = 1; d < 16; d <<= 1) rsum += __shfl_xor(rsum, d, 32);
      lrow[v] = lrow[v] * alpha + rsum;
#pragma unroll
      for (int t4 = 0; t4 < 4; ++t4) acc[t4][v] *= alpha;
      p0[v] = e0; p1[v] = e1;
    }

    // ---- bounce P through LDS: C-layout -> A-layout (16x32 f16, 1 KB)
#pragma unroll
    for (int v = 0; v < 8; ++v) {
      int rr = v + (hi << 3);
      pshm[rr * 32 + lo16]      = (_Float16)p0[v];
      pshm[rr * 32 + 16 + lo16] = (_Float16)p1[v];
    }
    __syncthreads();
    v16h pa = load_afrag(&pshm[lo16 * 32], khalf);

    // ---- O += P @ V  (B-frags contiguous over keys via V^T)
#pragma unroll
    for (int t4 = 0; t4 < 4; ++t4) {
      const _Float16* vp =
          vT + (size_t)(b * HIDDEN + h * HEAD_DIM + t4 * 16 + lo16) * SEQ + kb + kq;
      acc[t4] = __builtin_amdgcn_wmma_f32_16x16x32_f16(false, pa, false, load_bfrag(vp),
                                                       (short)0, acc[t4], false, false);
    }
    __syncthreads();   // protect pshm before next iteration overwrites it
  }

  // ---- finalize: divide by row sums, store f16 in [B,S,H*D] for out-proj
#pragma unroll
  for (int v = 0; v < 8; ++v) {
    float inv = 1.0f / lrow[v];
    int row   = b * SEQ + qbase + v + (hi << 3);
#pragma unroll
    for (int t4 = 0; t4 < 4; ++t4)
      aout[(size_t)row * HIDDEN + h * HEAD_DIM + t4 * 16 + lo16] =
          (_Float16)(acc[t4][v] * inv);
  }
}

// ---------------------------------------------------------------------------
// Launcher
// ---------------------------------------------------------------------------
extern "C" void kernel_launch(void* const* d_in, const int* in_sizes, int n_in,
                              void* d_out, int out_size, void* d_ws, size_t ws_size,
                              hipStream_t stream) {
  const float* x  = (const float*)d_in[0];
  const int*   am = (const int*)d_in[1];
  const float* wq = (const float*)d_in[2];
  const float* wk = (const float*)d_in[3];
  const float* wv = (const float*)d_in[4];
  const float* wo = (const float*)d_in[5];
  const float* bo = (const float*)d_in[6];
  const float* sq = (const float*)d_in[7];
  const float* sk = (const float*)d_in[8];
  const float* sv = (const float*)d_in[9];
  const float* so = (const float*)d_in[10];

  char* ws = (char*)d_ws;
  _Float16* x16  = (_Float16*)(ws + ((size_t)0  << 20));
  _Float16* wq16 = (_Float16*)(ws + ((size_t)8  << 20));
  _Float16* wk16 = (_Float16*)(ws + ((size_t)10 << 20));
  _Float16* wv16 = (_Float16*)(ws + ((size_t)12 << 20));
  _Float16* wo16 = (_Float16*)(ws + ((size_t)14 << 20));
  _Float16* q16  = (_Float16*)(ws + ((size_t)16 << 20));
  _Float16* k16  = (_Float16*)(ws + ((size_t)24 << 20));
  _Float16* v16  = (_Float16*)(ws + ((size_t)32 << 20));
  _Float16* vT16 = (_Float16*)(ws + ((size_t)40 << 20));
  _Float16* a16  = (_Float16*)(ws + ((size_t)48 << 20));
  unsigned* mpk  = (unsigned*)(ws + ((size_t)56 << 20));   // 512 KB packed mask

  const int nx = NROWS * HIDDEN;           // 4 M elements
  cvt_f32_f16<<<nx / 256, 256, 0, stream>>>(x, x16, nx);
  prep_weight<<<4096, 256, 0, stream>>>(wq, sq, wq16);
  prep_weight<<<4096, 256, 0, stream>>>(wk, sk, wk16);
  prep_weight<<<4096, 256, 0, stream>>>(wv, sv, wv16);
  prep_weight<<<4096, 256, 0, stream>>>(wo, so, wo16);
  pack_mask<<<(SEQ * SEQ) / 256, 256, 0, stream>>>(am, mpk);

  // blocks = (M/128) * (N/64) = 32 * 16 = 512, 256 threads each
  gemm_wmma<false><<<512, 256, 0, stream>>>(x16, wq16, nullptr, q16, NROWS, HIDDEN);
  gemm_wmma<false><<<512, 256, 0, stream>>>(x16, wk16, nullptr, k16, NROWS, HIDDEN);
  gemm_wmma<false><<<512, 256, 0, stream>>>(x16, wv16, nullptr, v16, NROWS, HIDDEN);

  transpose_v<<<nx / 256, 256, 0, stream>>>(v16, vT16);

  attn_wmma<<<BATCH * HEADS * (SEQ / 16), 32, 0, stream>>>(q16, k16, vT16, mpk, a16);

  gemm_wmma<true><<<512, 256, 0, stream>>>(a16, wo16, bo, d_out, NROWS, HIDDEN);
}
